// retina_polar2_scale_89017492177071
// MI455X (gfx1250) — compile-verified, matrix operand order checked by reading, not run on hardware
//
#include <hip/hip_runtime.h>
#include <hip/hip_bf16.h>

typedef __attribute__((ext_vector_type(16))) _Float16 v16h;
typedef __attribute__((ext_vector_type(8)))  _Float16 v8h;
typedef __attribute__((ext_vector_type(2)))  _Float16 v2h;
typedef __attribute__((ext_vector_type(8)))  float    v8f;

#define BATCH 128
#define IMG   224

#define SHUF16(lo, hi) __builtin_shufflevector(lo, hi, 0,1,2,3,4,5,6,7,8,9,10,11,12,13,14,15)

// ---------------------------------------------------------------------------
// K0: zero conv2 channel-sum accumulator + pack conv weights into zero-padded
// f16 im2col B-matrices:  w1p[16][96]  (n<6,k<75 used),  w2p[16][160] (k<150)
// k = c*25 + ky*5 + kx
// ---------------------------------------------------------------------------
__global__ __launch_bounds__(256) void k0_prep(
    const float* __restrict__ w1, const float* __restrict__ w2,
    _Float16* __restrict__ w1p, _Float16* __restrict__ w2p,
    float* __restrict__ gsums)
{
  int t = blockIdx.x * 256 + threadIdx.x;
  if (t < BATCH * 16) gsums[t] = 0.f;
  if (t < 16 * 96) {
    int n = t / 96, k = t % 96;
    _Float16 v = (_Float16)0.f;
    if (n < 6 && k < 75) {
      int c = k / 25, rem = k % 25, ky = rem / 5, kx = rem % 5;
      v = (_Float16)w1[((n * 3 + c) * 5 + ky) * 5 + kx];
    }
    w1p[t] = v;
  }
  if (t < 16 * 160) {
    int n = t / 160, k = t % 160;
    _Float16 v = (_Float16)0.f;
    if (k < 150) {
      int c = k / 25, rem = k % 25, ky = rem / 5, kx = rem % 5;
      v = (_Float16)w2[((n * 6 + c) * 5 + ky) * 5 + kx];
    }
    w2p[t] = v;
  }
}

// ---------------------------------------------------------------------------
// K1: conv1 (3->6, 5x5 valid) + 2x2 maxpool + bias -> f16 [128,6,110,110].
// One wave: 8 pooled x (16 conv x) of one pooled row, 6 channels.
// im2col staged in LDS (row stride 104 halves = 208B, 16B aligned), fragments
// are two ds_load_b128 each; B fragments are two global_load_b128 from w1p.
// All k->(c,ky,kx) decoding is compile-time (uniform unrolled k loop).
// ---------------------------------------------------------------------------
__global__ __launch_bounds__(128) void k1_conv1_pool(
    const float* __restrict__ x, const _Float16* __restrict__ w1p,
    const float* __restrict__ b1, _Float16* __restrict__ a1)
{
  __shared__ _Float16 A[4][32 * 104];       // per wave: rows (r*16+m), K pad 96
  const int warp = threadIdx.x >> 5;
  const int lane = threadIdx.x & 31;
  const int gwave = blockIdx.x * 4 + warp;  // grid exact: 128*110*14 waves
  const int seg = gwave % 14;
  int t = gwave / 14;
  const int py = t % 110;
  const int b  = t / 110;
  const int x0 = seg * 16, y0 = py * 2;

  const int half = lane >> 4;               // = conv row r during build
  const int m    = lane & 15;

  // ---- build im2col row (r=half, m): k compile-time in every iteration ----
  {
    const float* xrow = x + (size_t)b * 3 * 50176 + (size_t)(y0 + half) * 224;
    const int xm = x0 + m;
    _Float16* Arow = &A[warp][lane * 104];
    #pragma unroll
    for (int k2 = 0; k2 < 48; ++k2) {
      v2h p;
      #pragma unroll
      for (int j = 0; j < 2; ++j) {
        const int k = k2 * 2 + j;
        _Float16 v = (_Float16)0.f;
        if (k < 75) {                       // compile-time predicate
          const int c = k / 25, rem = k % 25, ky = rem / 5, kx = rem % 5;
          int gx = min(xm + kx, IMG - 1);   // border clamp (invalid px discarded)
          v = (_Float16)xrow[c * 50176 + ky * 224 + gx];
        }
        p[j] = v;
      }
      *(v2h*)(Arow + k2 * 2) = p;           // ds_store_b32
    }
  }

  // ---- B fragments: two 16B global loads each from packed weights ----
  const _Float16* Wn = w1p + (size_t)m * 96;     // n == lane&15
  v16h bf[3];
  #pragma unroll
  for (int kk = 0; kk < 3; ++kk) {
    v8h lo = *(const v8h*)(Wn + kk * 32 + half * 8);
    v8h hi = *(const v8h*)(Wn + kk * 32 + 16 + half * 8);
    bf[kk] = SHUF16(lo, hi);
  }

  // ---- two conv rows of WMMA; A fragments: two ds_load_b128 each ----
  v8f acc[2];
  #pragma unroll
  for (int r = 0; r < 2; ++r) {
    const _Float16* Ar = &A[warp][(r * 16 + m) * 104];
    v8f c8 = {0.f, 0.f, 0.f, 0.f, 0.f, 0.f, 0.f, 0.f};
    #pragma unroll
    for (int kk = 0; kk < 3; ++kk) {
      v8h lo = *(const v8h*)(Ar + kk * 32 + half * 8);
      v8h hi = *(const v8h*)(Ar + kk * 32 + 16 + half * 8);
      v16h a = SHUF16(lo, hi);
      c8 = __builtin_amdgcn_wmma_f32_16x16x32_f16(false, a, false, bf[kk],
                                                  (short)0, c8, false, false);
    }
    acc[r] = c8;
  }

  // ---- 2x2 maxpool in registers + bias; D: N=lane%16, M=(half?8:0)+v ----
  if (m < 6) {
    float bias = b1[m];
    #pragma unroll
    for (int p = 0; p < 4; ++p) {
      float r0 = fmaxf(acc[0][2 * p],     acc[1][2 * p]);
      float r1 = fmaxf(acc[0][2 * p + 1], acc[1][2 * p + 1]);
      float pv = fmaxf(r0, r1) + bias;
      int px = seg * 8 + half * 4 + p;
      if (px < 110)
        a1[((b * 6 + m) * 110 + py) * 110 + px] = (_Float16)pv;
    }
  }
}

// ---------------------------------------------------------------------------
// K2: conv2 (6->16, 5x5 valid) on f16 glimpse + fused global channel sum.
// M=16 spatial, N=16 channels, K=150 pad 160 (5 WMMAs). im2col in LDS
// (row stride 168 halves = 336B); half-waves split K range so k is
// compile-time in both branch bodies. grid = 128 * 186 blocks of 4 waves.
// ---------------------------------------------------------------------------
__global__ __launch_bounds__(128) void k2_conv2_sum(
    const _Float16* __restrict__ a1, const _Float16* __restrict__ w2p,
    float* __restrict__ gsums)
{
  __shared__ _Float16 A[4][16 * 168];
  __shared__ float bsum[16];
  const int warp = threadIdx.x >> 5;
  const int lane = threadIdx.x & 31;
  if (threadIdx.x < 16) bsum[threadIdx.x] = 0.f;
  __syncthreads();

  const int b    = blockIdx.x / 186;
  const int blk  = blockIdx.x % 186;
  const int widx = blk * 4 + warp;
  const bool active = (widx < 742);         // 106 rows x 7 x-segments
  const int widx_c = min(widx, 741);        // inactive waves run same path
  const int y    = widx_c / 7;
  const int xseg = widx_c % 7;
  const int x0   = xseg * 16;

  const int half = lane >> 4;
  const int m    = lane & 15;

  // ---- build im2col row m; lanes<16: k in [0,80), lanes>=16: k in [80,160) ----
  {
    const _Float16* src = a1 + (size_t)b * 6 * 12100 + (size_t)y * 110;
    const int xm = x0 + m;
    _Float16* Arow = &A[warp][m * 168];
    if (half == 0) {
      #pragma unroll
      for (int k2 = 0; k2 < 40; ++k2) {
        v2h p;
        #pragma unroll
        for (int j = 0; j < 2; ++j) {
          const int k = k2 * 2 + j;
          const int c = k / 25, rem = k % 25, ky = rem / 5, kx = rem % 5;
          int gx = min(xm + kx, 109);
          p[j] = src[(size_t)c * 12100 + ky * 110 + gx];
        }
        *(v2h*)(Arow + k2 * 2) = p;
      }
    } else {
      #pragma unroll
      for (int k2 = 0; k2 < 40; ++k2) {
        v2h p;
        #pragma unroll
        for (int j = 0; j < 2; ++j) {
          const int k = 80 + k2 * 2 + j;
          _Float16 v = (_Float16)0.f;
          if (k < 150) {
            const int c = k / 25, rem = k % 25, ky = rem / 5, kx = rem % 5;
            int gx = min(xm + kx, 109);
            v = src[(size_t)c * 12100 + ky * 110 + gx];
          }
          p[j] = v;
        }
        *(v2h*)(Arow + 80 + k2 * 2) = p;
      }
    }
  }

  // ---- 5 WMMA steps; A from LDS b128 pairs, B from packed global weights ----
  const _Float16* Wn = w2p + (size_t)m * 160;
  const _Float16* Ar = &A[warp][m * 168];
  v8f acc = {0.f, 0.f, 0.f, 0.f, 0.f, 0.f, 0.f, 0.f};
  #pragma unroll
  for (int kk = 0; kk < 5; ++kk) {
    v8h blo = *(const v8h*)(Wn + kk * 32 + half * 8);
    v8h bhi = *(const v8h*)(Wn + kk * 32 + 16 + half * 8);
    v16h bfr = SHUF16(blo, bhi);
    v8h alo = *(const v8h*)(Ar + kk * 32 + half * 8);
    v8h ahi = *(const v8h*)(Ar + kk * 32 + 16 + half * 8);
    v16h a = SHUF16(alo, ahi);
    acc = __builtin_amdgcn_wmma_f32_16x16x32_f16(false, a, false, bfr,
                                                 (short)0, acc, false, false);
  }

  // ---- fold spatially-valid outputs into per-channel partial sums ----
  float lsum = 0.f;
  if (active) {
    #pragma unroll
    for (int v = 0; v < 8; ++v) {
      int xx = x0 + half * 8 + v;
      if (xx < 106) lsum += acc[v];
    }
  }
  lsum += __shfl_xor(lsum, 16, 32);         // combine half-rows (same channel)
  if (active && lane < 16) atomicAdd(&bsum[lane], lsum);
  __syncthreads();
  if (threadIdx.x < 16) atomicAdd(&gsums[b * 16 + threadIdx.x], bsum[threadIdx.x]);
}

// ---------------------------------------------------------------------------
// K3: mean + bias -> fc1(relu) -> fc2 -> 2*sigmoid -> weight into d_out[0:256]
// ---------------------------------------------------------------------------
__global__ void k3_fc(const float* __restrict__ gsums, const float* __restrict__ b2,
                      const float* __restrict__ f1w, const float* __restrict__ f1b,
                      const float* __restrict__ f2w, const float* __restrict__ f2b,
                      float* __restrict__ out)
{
  int b = threadIdx.x;
  if (b >= BATCH) return;
  const float inv = 1.f / (106.f * 106.f);
  float mean[16];
  #pragma unroll
  for (int c = 0; c < 16; ++c) mean[c] = gsums[b * 16 + c] * inv + b2[c];
  float h[8];
  #pragma unroll
  for (int j = 0; j < 8; ++j) {
    float s = f1b[j];
    #pragma unroll
    for (int c = 0; c < 16; ++c) s += f1w[j * 16 + c] * mean[c];
    h[j] = fmaxf(s, 0.f);
  }
  #pragma unroll
  for (int o = 0; o < 2; ++o) {
    float s = f2b[o];
    #pragma unroll
    for (int j = 0; j < 8; ++j) s += f2w[o * 8 + j] * h[j];
    out[b * 2 + o] = 2.0f / (1.f + __expf(-s));   // ALPHA * sigmoid
  }
}

// ---------------------------------------------------------------------------
// K4: attention-scaled polar grid sample (border bilinear) + 10x10 avg pool.
// One wave per pooled cell (65536 waves), 100 samples / 32 lanes.
// ---------------------------------------------------------------------------
__global__ __launch_bounds__(256) void k4_sample_pool(
    const float* __restrict__ x, const float* __restrict__ lt,
    const float* __restrict__ grid2d, const float* wout, float* out)
{
  const int lane  = threadIdx.x & 31;
  const int gwave = blockIdx.x * 8 + (threadIdx.x >> 5);
  const int cx = gwave % 32;
  int t = gwave / 32;
  const int cy = t % 16;
  const int b  = t / 16;

  const float w0 = wout[b * 2 + 0], w1v = wout[b * 2 + 1];
  const float lx = lt[b * 2 + 0],  ly  = lt[b * 2 + 1];
  const float* img = x + (size_t)b * 3 * IMG * IMG;

  float s0 = 0.f, s1 = 0.f, s2 = 0.f;
  for (int s = lane; s < 100; s += 32) {
    int sy = s / 10, sx = s % 10;
    int gy = cy * 10 + sy, gx = cx * 10 + sx;
    const float g0 = grid2d[(gy * 320 + gx) * 2 + 0];
    const float g1 = grid2d[(gy * 320 + gx) * 2 + 1];
    float fx = (lx + g0) * w0, fy = (ly + g1) * w1v;
    float ix = fminf(fmaxf(((fx + 1.f) * 224.f - 1.f) * 0.5f, 0.f), 223.f);
    float iy = fminf(fmaxf(((fy + 1.f) * 224.f - 1.f) * 0.5f, 0.f), 223.f);
    float fx0 = floorf(ix), fy0 = floorf(iy);
    int x0 = (int)fx0, y0 = (int)fy0;
    int x1 = min(x0 + 1, 223), y1 = min(y0 + 1, 223);
    float wx = ix - fx0, wy = iy - fy0;
    float w00 = (1.f - wx) * (1.f - wy), w01 = wx * (1.f - wy);
    float w10 = (1.f - wx) * wy,         w11 = wx * wy;
    int o00 = y0 * IMG + x0, o01 = y0 * IMG + x1;
    int o10 = y1 * IMG + x0, o11 = y1 * IMG + x1;
    s0 += img[o00] * w00 + img[o01] * w01 + img[o10] * w10 + img[o11] * w11;
    const float* i1 = img + IMG * IMG;
    s1 += i1[o00] * w00 + i1[o01] * w01 + i1[o10] * w10 + i1[o11] * w11;
    const float* i2 = img + 2 * IMG * IMG;
    s2 += i2[o00] * w00 + i2[o01] * w01 + i2[o10] * w10 + i2[o11] * w11;
  }
  #pragma unroll
  for (int off = 16; off; off >>= 1) {
    s0 += __shfl_down(s0, off, 32);
    s1 += __shfl_down(s1, off, 32);
    s2 += __shfl_down(s2, off, 32);
  }
  if (lane == 0) {
    float* p = out + 256;                  // pooled region after weight[128,2]
    p[((b * 3 + 0) * 16 + cy) * 32 + cx] = s0 * 0.01f;
    p[((b * 3 + 1) * 16 + cy) * 32 + cx] = s1 * 0.01f;
    p[((b * 3 + 2) * 16 + cy) * 32 + cx] = s2 * 0.01f;
  }
}

// ---------------------------------------------------------------------------
extern "C" void kernel_launch(void* const* d_in, const int* in_sizes, int n_in,
                              void* d_out, int out_size, void* d_ws, size_t ws_size,
                              hipStream_t stream) {
  const float* x    = (const float*)d_in[0];
  const float* lt   = (const float*)d_in[1];
  const float* grid = (const float*)d_in[2];
  const float* w1   = (const float*)d_in[3];
  const float* b1   = (const float*)d_in[4];
  const float* w2   = (const float*)d_in[5];
  const float* b2   = (const float*)d_in[6];
  const float* f1w  = (const float*)d_in[7];
  const float* f1b  = (const float*)d_in[8];
  const float* f2w  = (const float*)d_in[9];
  const float* f2b  = (const float*)d_in[10];
  float* out = (float*)d_out;

  // ws layout (16B-aligned slabs):
  //   a1    f16 [128,6,110,110]  = 18,585,600 B
  //   gsums f32 [128,16]         =      8,192 B
  //   w1p   f16 [16][96]         =      3,072 B
  //   w2p   f16 [16][160]        =      5,120 B
  char* wsb = (char*)d_ws;
  _Float16* a1   = (_Float16*)(wsb);
  float*    gsums = (float*)(wsb + 18585600u);
  _Float16* w1p  = (_Float16*)(wsb + 18585600u + 8192u);
  _Float16* w2p  = (_Float16*)(wsb + 18585600u + 8192u + 3072u);

  k0_prep<<<10, 256, 0, stream>>>(w1, w2, w1p, w2p, gsums);
  k1_conv1_pool<<<(BATCH * 110 * 14) / 4, 128, 0, stream>>>(x, w1p, b1, a1);
  k2_conv2_sum<<<BATCH * 186, 128, 0, stream>>>(a1, w2p, gsums);
  k3_fc<<<1, 128, 0, stream>>>(gsums, b2, f1w, f1b, f2w, f2b, out);
  k4_sample_pool<<<(BATCH * 16 * 32) / 8, 256, 0, stream>>>(x, lt, grid, out, out);
}
